// SelfAttention_7103875908230
// MI455X (gfx1250) — compile-verified, matrix operand order checked by reading, not run on hardware
//
#include <hip/hip_runtime.h>
#include <hip/hip_fp16.h>

typedef __attribute__((ext_vector_type(16))) _Float16 v16h;
typedef __attribute__((ext_vector_type(8)))  float    v8f;

namespace {

constexpr int BB = 8;     // batch
constexpr int TT = 2048;  // sequence
constexpr int CC = 1024;  // embed dim
constexpr int HH = 1024;  // head size

constexpr int TM = 128;       // output tile rows per block
constexpr int TN = 128;       // output tile cols per block
constexpr int TK = 32;        // k-chunk (matches wmma 16x16x32)
// 40 halves = 80 bytes: 16B-aligned rows (needed by async b128 LDS fills) and
// conflict-free fragment reads (bank = 20*m mod 64 is distinct for m=0..15).
constexpr int LSTR = TK + 8;

union HalfPack { uint4 q; _Float16 h[8]; };

// CDNA5 async memory->LDS copy (16 bytes per lane), tracked on ASYNCcnt.
// No VGPR round-trip: the DMA writes LDS directly.
__device__ __forceinline__ void async_copy16(const void* gsrc, void* ldst) {
  const unsigned lds = (unsigned)(size_t)ldst;           // low 32 bits = LDS byte addr
  const unsigned long long ga = (unsigned long long)(size_t)gsrc;
  asm volatile("global_load_async_to_lds_b128 %0, %1, off"
               :: "v"(lds), "v"(ga)
               : "memory");
}
__device__ __forceinline__ void wait_async() {
  asm volatile("s_wait_asynccnt 0x0" ::: "memory");
}

// Load a 16x32 (A) or 32x16-as-[n][k] (B) fp16 fragment from an LDS tile laid
// out [row][k] with stride LSTR. Follows the CDNA5 16-bit A-matrix lane map:
//   lane&15 -> row, lane>>4 -> K-half offset (0/8), dword j -> packed K pair.
__device__ __forceinline__ v16h frag_ld(const _Float16* lds, int rowBase) {
  const int lane = threadIdx.x & 31;
  const int r  = rowBase + (lane & 15);
  const int kb = (lane >> 4) * 8;
  v16h f;
  unsigned int* fu = reinterpret_cast<unsigned int*>(&f);
  const _Float16* rp = lds + r * LSTR;
#pragma unroll
  for (int j = 0; j < 8; ++j) {
    const int k = ((j & 3) * 2) + ((j >> 2) * 16) + kb;  // 0,2,4,6,16,18,20,22
    fu[j] = *reinterpret_cast<const unsigned int*>(rp + k);
  }
  return f;
}

__device__ __forceinline__ v8f wmma16(const v16h& a, const v16h& b, const v8f& c) {
  return __builtin_amdgcn_wmma_f32_16x16x32_f16(
      /*neg_a=*/false, a, /*neg_b=*/false, b,
      /*c_mod=*/(short)0, c, /*reuse_a=*/false, /*reuse_b=*/false);
}

} // namespace

// ---------------------------------------------------------------------------
// Kernel 1: Q/K/V = x @ W{q,k,v}  (fp32 in, fp16 out). grid.z selects matrix.
// Staging converts f32->f16, so it must flow through VGPRs (no async path).
// ---------------------------------------------------------------------------
__global__ void __launch_bounds__(256)
qkv_gemm(const float* __restrict__ x,
         const float* __restrict__ Wq,
         const float* __restrict__ Wk,
         const float* __restrict__ Wv,
         _Float16* __restrict__ Qo,
         _Float16* __restrict__ Ko,
         _Float16* __restrict__ Vo) {
  __shared__ _Float16 lA[TM * LSTR];
  __shared__ _Float16 lB[TN * LSTR];

  const int m0 = blockIdx.x * TM;  // row in flattened [B*T]
  const int n0 = blockIdx.y * TN;  // col in [H]
  const float* W = (blockIdx.z == 0) ? Wq : (blockIdx.z == 1) ? Wk : Wv;
  _Float16* Out  = (blockIdx.z == 0) ? Qo : (blockIdx.z == 1) ? Ko : Vo;

  const int tid = threadIdx.x;
  const int wid = tid >> 5;   // 8 waves of 32
  const int mb  = wid * 16;   // wave's 16-row strip

  v8f acc[8] = {};

  for (int kc = 0; kc < CC / TK; ++kc) {
    // Stage A: x[128 x 32] fp32 -> fp16, row-major [m][k]
    {
      const int row = tid >> 1;
      const int cb  = (tid & 1) * 16;
      const float* src = x + (size_t)(m0 + row) * CC + kc * TK + cb;
      const float4* g = reinterpret_cast<const float4*>(src);
#pragma unroll
      for (int i = 0; i < 4; ++i) {
        float4 v = g[i];
        _Float16* d = &lA[row * LSTR + cb + i * 4];
        d[0] = (_Float16)v.x; d[1] = (_Float16)v.y;
        d[2] = (_Float16)v.z; d[3] = (_Float16)v.w;
      }
      if (kc + 1 < CC / TK)
        __builtin_prefetch(src + TK, 0, 0);
    }
    // Stage B transposed: W[32 k x 128 n] fp32 -> lB[n][k] fp16
    {
      const int kr  = tid >> 3;        // 0..31
      const int nb4 = (tid & 7) * 16;  // 0..112
      const float* src = W + (size_t)(kc * TK + kr) * HH + n0 + nb4;
      const float4* g = reinterpret_cast<const float4*>(src);
#pragma unroll
      for (int i = 0; i < 4; ++i) {
        float4 v = g[i];
        lB[(nb4 + i * 4 + 0) * LSTR + kr] = (_Float16)v.x;
        lB[(nb4 + i * 4 + 1) * LSTR + kr] = (_Float16)v.y;
        lB[(nb4 + i * 4 + 2) * LSTR + kr] = (_Float16)v.z;
        lB[(nb4 + i * 4 + 3) * LSTR + kr] = (_Float16)v.w;
      }
      if (kc + 1 < CC / TK)
        __builtin_prefetch(src + (size_t)TK * HH, 0, 0);
    }
    __syncthreads();
    v16h a = frag_ld(lA, mb);
#pragma unroll
    for (int nb = 0; nb < 8; ++nb) {
      v16h bf = frag_ld(lB, nb * 16);
      acc[nb] = wmma16(a, bf, acc[nb]);
    }
    __syncthreads();
  }

  const int lane = tid & 31;
  const int ln = lane & 15;
  const int mh = (lane >> 4) * 8;
#pragma unroll
  for (int nb = 0; nb < 8; ++nb)
#pragma unroll
    for (int r = 0; r < 8; ++r) {
      const int m = m0 + mb + mh + r;
      const int n = n0 + nb * 16 + ln;
      Out[(size_t)m * HH + n] = (_Float16)acc[nb][r];
    }
}

// ---------------------------------------------------------------------------
// Kernel 2: P = (Q @ K^T) / sqrt(C), causal mask -> -inf, fp16 out.
// K stored [s][h] row-major IS B^T for this GEMM, and both tiles are pure
// fp16 copies -> use async global->LDS DMA (ASYNCcnt) for the fills.
// ---------------------------------------------------------------------------
__global__ void __launch_bounds__(256)
qk_scores(const _Float16* __restrict__ Q,
          const _Float16* __restrict__ Kf,
          _Float16* __restrict__ P,
          const int* __restrict__ maskFlag) {
  __shared__ _Float16 lA[TM * LSTR];
  __shared__ _Float16 lB[TN * LSTR];

  const int t0 = blockIdx.x * TM;
  const int s0 = blockIdx.y * TN;
  const int b  = blockIdx.z;
  const int tid = threadIdx.x;
  const bool masked = maskFlag[0] != 0;

  _Float16* Pb = P + (size_t)b * TT * TT;

  if (masked && s0 > t0 + (TM - 1)) {
    // Fully masked tile: fill with fp16 -inf, skip the GEMM.
    const int row = tid >> 1;
    const int cb  = (tid & 1) * 64;
    uint4* d = reinterpret_cast<uint4*>(Pb + (size_t)(t0 + row) * TT + s0 + cb);
    const uint4 ninf = make_uint4(0xFC00FC00u, 0xFC00FC00u, 0xFC00FC00u, 0xFC00FC00u);
#pragma unroll
    for (int i = 0; i < 8; ++i) d[i] = ninf;
    return;
  }

  const int wid = tid >> 5;
  const int mb  = wid * 16;
  v8f acc[8] = {};

  const _Float16* Qb = Q  + (size_t)b * TT * HH;
  const _Float16* Kb = Kf + (size_t)b * TT * HH;

  const int row = tid >> 1;
  const int cb  = (tid & 1) * 16;

  for (int kc = 0; kc < HH / TK; ++kc) {
    {  // Stage A: Q[128 t x 32 h] via async DMA (2 x 16B per thread)
      const _Float16* src = Qb + (size_t)(t0 + row) * HH + kc * TK + cb;
      _Float16* dst = &lA[row * LSTR + cb];
      async_copy16(src,     dst);
      async_copy16(src + 8, dst + 8);
      if (kc + 1 < HH / TK) __builtin_prefetch(src + TK, 0, 0);
    }
    {  // Stage B: K[128 s x 32 h] row-major == B^T, async direct copy
      const _Float16* src = Kb + (size_t)(s0 + row) * HH + kc * TK + cb;
      _Float16* dst = &lB[row * LSTR + cb];
      async_copy16(src,     dst);
      async_copy16(src + 8, dst + 8);
      if (kc + 1 < HH / TK) __builtin_prefetch(src + TK, 0, 0);
    }
    wait_async();
    __syncthreads();
    v16h a = frag_ld(lA, mb);
#pragma unroll
    for (int nb = 0; nb < 8; ++nb) {
      v16h bf = frag_ld(lB, nb * 16);
      acc[nb] = wmma16(a, bf, acc[nb]);
    }
    __syncthreads();
  }

  const int lane = tid & 31;
  const int ln = lane & 15;
  const int mh = (lane >> 4) * 8;
  const float scale = 0.03125f;  // 1/sqrt(C) = 1/32
#pragma unroll
  for (int nb = 0; nb < 8; ++nb)
#pragma unroll
    for (int r = 0; r < 8; ++r) {
      const int t = t0 + mb + mh + r;
      const int s = s0 + nb * 16 + ln;
      float v = acc[nb][r] * scale;
      if (masked && s > t) v = -__builtin_huge_valf();
      Pb[(size_t)t * TT + s] = (_Float16)v;
    }
}

// ---------------------------------------------------------------------------
// Kernel 3: in-place row softmax on P (fp16), one wave32 per row.
// ---------------------------------------------------------------------------
__global__ void __launch_bounds__(256)
softmax_rows(_Float16* __restrict__ P) {
  const int wid  = threadIdx.x >> 5;
  const int lane = threadIdx.x & 31;
  const size_t row = (size_t)blockIdx.x * 8 + wid;
  uint4* pv = reinterpret_cast<uint4*>(P + row * TT);

  float vals[64];
  float mx = -__builtin_huge_valf();
#pragma unroll
  for (int c = 0; c < 8; ++c) {
    HalfPack u; u.q = pv[c * 32 + lane];
#pragma unroll
    for (int j = 0; j < 8; ++j) {
      float v = (float)u.h[j];
      vals[c * 8 + j] = v;
      mx = fmaxf(mx, v);
    }
  }
#pragma unroll
  for (int off = 16; off > 0; off >>= 1)
    mx = fmaxf(mx, __shfl_xor(mx, off, 32));

  float sum = 0.f;
#pragma unroll
  for (int i = 0; i < 64; ++i) {
    float e = __expf(vals[i] - mx);  // exp(-inf)=0 handles the mask
    vals[i] = e;
    sum += e;
  }
#pragma unroll
  for (int off = 16; off > 0; off >>= 1)
    sum += __shfl_xor(sum, off, 32);
  const float inv = 1.f / sum;

#pragma unroll
  for (int c = 0; c < 8; ++c) {
    HalfPack u;
#pragma unroll
    for (int j = 0; j < 8; ++j) u.h[j] = (_Float16)(vals[c * 8 + j] * inv);
    pv[c * 32 + lane] = u.q;
  }
}

// ---------------------------------------------------------------------------
// Kernel 4: O = P @ V (fp16 x fp16 -> fp32 out). Causal truncates the k-loop.
// A tile (P) is a pure copy -> async DMA; B tile (V) is transposed in LDS.
// ---------------------------------------------------------------------------
__global__ void __launch_bounds__(256)
pv_gemm(const _Float16* __restrict__ P,
        const _Float16* __restrict__ V,
        float* __restrict__ O,
        const int* __restrict__ maskFlag) {
  __shared__ _Float16 lA[TM * LSTR];
  __shared__ _Float16 lB[TN * LSTR];

  const int t0 = blockIdx.x * TM;
  const int h0 = blockIdx.y * TN;
  const int b  = blockIdx.z;
  const int tid = threadIdx.x;
  const int wid = tid >> 5;
  const int mb  = wid * 16;

  const _Float16* Pb = P + (size_t)b * TT * TT;
  const _Float16* Vb = V + (size_t)b * TT * HH;

  const bool masked = maskFlag[0] != 0;
  const int nk = masked ? (t0 / TK + TM / TK) : (TT / TK);  // rows s<=t0+127 only

  v8f acc[8] = {};
  for (int kc = 0; kc < nk; ++kc) {
    {  // Stage A: P[128 t x 32 s] via async DMA
      const int row = tid >> 1;
      const int cb  = (tid & 1) * 16;
      const _Float16* src = Pb + (size_t)(t0 + row) * TT + kc * TK + cb;
      _Float16* dst = &lA[row * LSTR + cb];
      async_copy16(src,     dst);
      async_copy16(src + 8, dst + 8);
      if (kc + 1 < nk) __builtin_prefetch(src + TK, 0, 0);
    }
    {  // Stage B transposed: V[32 s x 128 h] -> lB[h][s] (VGPR path)
      const int sr = tid >> 3;        // 0..31
      const int hb = (tid & 7) * 16;  // 0..112
      const _Float16* src = Vb + (size_t)(kc * TK + sr) * HH + h0 + hb;
      const uint4* g = reinterpret_cast<const uint4*>(src);
      HalfPack a0, a1; a0.q = g[0]; a1.q = g[1];
#pragma unroll
      for (int i = 0; i < 8; ++i) {
        lB[(hb + i)     * LSTR + sr] = a0.h[i];
        lB[(hb + 8 + i) * LSTR + sr] = a1.h[i];
      }
      if (kc + 1 < nk) __builtin_prefetch(src + (size_t)TK * HH, 0, 0);
    }
    wait_async();
    __syncthreads();
    v16h a = frag_ld(lA, mb);
#pragma unroll
    for (int nb = 0; nb < 8; ++nb) {
      v16h bf = frag_ld(lB, nb * 16);
      acc[nb] = wmma16(a, bf, acc[nb]);
    }
    __syncthreads();
  }

  const int lane = tid & 31;
  const int ln = lane & 15;
  const int mh = (lane >> 4) * 8;
#pragma unroll
  for (int nb = 0; nb < 8; ++nb)
#pragma unroll
    for (int r = 0; r < 8; ++r) {
      const int t = t0 + mb + mh + r;
      const int h = h0 + nb * 16 + ln;
      O[((size_t)b * TT + t) * HH + h] = acc[nb][r];
    }
}

// ---------------------------------------------------------------------------
extern "C" void kernel_launch(void* const* d_in, const int* in_sizes, int n_in,
                              void* d_out, int out_size, void* d_ws, size_t ws_size,
                              hipStream_t stream) {
  const float* x  = (const float*)d_in[0];
  const float* Wq = (const float*)d_in[1];
  const float* Wk = (const float*)d_in[2];
  const float* Wv = (const float*)d_in[3];
  const int*   mk = (const int*)d_in[4];
  float* out = (float*)d_out;

  // Workspace layout (fp16): Q | K | V (32 MB each) | P (64 MB) = 160 MB.
  const size_t qkvElems = (size_t)BB * TT * HH;
  _Float16* Q  = (_Float16*)d_ws;
  _Float16* Kf = Q  + qkvElems;
  _Float16* V  = Kf + qkvElems;
  _Float16* P  = V  + qkvElems;

  const dim3 blk(256, 1, 1);
  qkv_gemm<<<dim3((BB * TT) / TM, HH / TN, 3), blk, 0, stream>>>(x, Wq, Wk, Wv, Q, Kf, V);
  qk_scores<<<dim3(TT / TM, TT / TN, BB), blk, 0, stream>>>(Q, Kf, P, mk);
  softmax_rows<<<dim3((BB * TT) / 8, 1, 1), blk, 0, stream>>>(P);
  pv_gemm<<<dim3(TT / TM, HH / TN, BB), blk, 0, stream>>>(P, V, out, mk);
}
